// Encoder_30021821399397
// MI455X (gfx1250) — matrix operand order repaired and checked by reading, not compiled, not run on hardware
//
#include <hip/hip_runtime.h>

// ---------------------------------------------------------------------------
// Problem constants (match reference): L=4, B=2, S=2048, D=1024, HID=1024,
// NH=16, C=64.  All GEMMs are [4096,1024] x [1024,1024].
// ---------------------------------------------------------------------------

#define TB 256
#define BM 256          // block tile rows
#define BN 128          // block tile cols
#define BKK 32          // k-step (one WMMA depth)
#define LDS_STRIDE 40   // 32 elements + 8 pad (80 bytes, 16B-aligned rows)

typedef __attribute__((ext_vector_type(16))) __bf16        v16bf;
typedef __attribute__((ext_vector_type(8)))  float         v8f;
typedef __attribute__((ext_vector_type(4)))  unsigned int  v4u;

union Frag { v4u u[2]; v16bf v; };

__device__ __forceinline__ unsigned short f2bf(float f) {
  unsigned int u = __float_as_uint(f);
  u += 0x7FFFu + ((u >> 16) & 1u);          // round-to-nearest-even
  return (unsigned short)(u >> 16);
}

// gfx1250 async copy: global memory -> LDS, 16 bytes per lane, ASYNCcnt-tracked.
// GV addressing: vdst = LDS byte offset (low 32 bits of generic LDS pointer),
// vaddr = 64-bit global address, saddr = off (NULL).
__device__ __forceinline__ void async_copy_b128(unsigned lds_off, const void* gptr) {
  asm volatile("global_load_async_to_lds_b128 %0, %1, off"
               :: "v"(lds_off), "v"(gptr) : "memory");
}
__device__ __forceinline__ void wait_async0() {
  asm volatile("s_wait_asynccnt 0x0" ::: "memory");
}

// ---------------------------------------------------------------------------
// WMMA GEMM:  C[M,N] = act( (A_bf16[M,K] @ Bt_bf16[N,K]^T + bias[N]) * scale )
// A row-major [M,K] bf16, Bt row-major [N,K] bf16 (i.e. W transposed).
// 256 threads / 8 waves; block tile 256x128; wave tile 64x64 (4x4 WMMA accs).
// Double-buffered LDS fed by GLOBAL_LOAD_ASYNC_TO_LDS_B128.
// ---------------------------------------------------------------------------
template<bool RELU, bool WF32, bool WBF16>
__global__ __launch_bounds__(TB)
void gemm_bf16_wmma(const unsigned short* __restrict__ A,
                    const unsigned short* __restrict__ Bt,
                    const float* __restrict__ bias,
                    float postScale,
                    float* __restrict__ outF,
                    unsigned short* __restrict__ outB,
                    int M, int N, int K)
{
  __shared__ alignas(16) unsigned short As[2][BM * LDS_STRIDE];
  __shared__ alignas(16) unsigned short Bs[2][BN * LDS_STRIDE];

  const int tid  = threadIdx.x;
  const int lane = tid & 31;
  const int wid  = tid >> 5;
  const int wm   = wid & 3;       // 4 wave-rows  (4 x 64 rows)
  const int wn   = wid >> 2;      // 2 wave-cols  (2 x 64 cols)
  const int m15  = lane & 15;
  const int hi   = lane >> 4;

  const int rowBase = blockIdx.y * BM;
  const int colBase = blockIdx.x * BN;

  // cooperative tile copy: 4 x 16B chunks per 32-element row, 64 rows/pass
  const int ldRow   = tid >> 2;
  const int ldChunk = (tid & 3) * 8;

  v8f acc[4][4] = {};

  // issue async copies for the k0 tile into LDS buffer p
  auto issue = [&](int k0, int p) {
    #pragma unroll
    for (int t = 0; t < 4; ++t) {                    // A: 256 rows
      const int row = ldRow + t * 64;
      const unsigned lofs =
          (unsigned)(unsigned long long)&As[p][row * LDS_STRIDE + ldChunk];
      async_copy_b128(lofs, A + (size_t)(rowBase + row) * K + k0 + ldChunk);
    }
    #pragma unroll
    for (int t = 0; t < 2; ++t) {                    // B: 128 rows
      const int row = ldRow + t * 64;
      const unsigned lofs =
          (unsigned)(unsigned long long)&Bs[p][row * LDS_STRIDE + ldChunk];
      async_copy_b128(lofs, Bt + (size_t)(colBase + row) * K + k0 + ldChunk);
    }
  };

  issue(0, 0);
  wait_async0();
  __syncthreads();

  int p = 0;
  for (int k0 = 0; k0 < K; k0 += BKK) {
    if (k0 + BKK < K) issue(k0 + BKK, p ^ 1);        // prefetch next tile (async)

    // B fragments: per-lane (col = m15), K = hi*16 + [0..15] contiguous
    Frag b[4];
    #pragma unroll
    for (int ni = 0; ni < 4; ++ni) {
      const unsigned short* q =
          &Bs[p][(wn * 64 + ni * 16 + m15) * LDS_STRIDE + hi * 16];
      b[ni].u[0] = *(const v4u*)q;
      b[ni].u[1] = *(const v4u*)(q + 8);
    }
    // A fragments: per-lane (row = m15), K = hi*16 + [0..15] contiguous
    #pragma unroll
    for (int mi = 0; mi < 4; ++mi) {
      Frag a;
      const unsigned short* q =
          &As[p][(wm * 64 + mi * 16 + m15) * LDS_STRIDE + hi * 16];
      a.u[0] = *(const v4u*)q;
      a.u[1] = *(const v4u*)(q + 8);
      #pragma unroll
      for (int ni = 0; ni < 4; ++ni)
        acc[mi][ni] = __builtin_amdgcn_wmma_f32_16x16x32_bf16(
            false, a.v, false, b[ni].v, (short)0, acc[mi][ni], false, false);
    }

    wait_async0();          // my async writes into buffer p^1 have landed
    __syncthreads();        // everyone done reading p / writing p^1
    p ^= 1;
  }

  // epilogue: C lane layout = row hi*8+r, col m15
  #pragma unroll
  for (int ni = 0; ni < 4; ++ni) {
    const int col  = colBase + wn * 64 + ni * 16 + m15;
    const float bc = bias[col];
    #pragma unroll
    for (int mi = 0; mi < 4; ++mi) {
      #pragma unroll
      for (int r = 0; r < 8; ++r) {
        const int row = rowBase + wm * 64 + mi * 16 + hi * 8 + r;
        float v = (acc[mi][ni][r] + bc) * postScale;
        if (RELU) v = fmaxf(v, 0.0f);
        const size_t o = (size_t)row * N + col;
        if (WF32)  outF[o] = v;
        if (WBF16) outB[o] = f2bf(v);
      }
    }
  }
}

// ---------------------------------------------------------------------------
// x + sinusoidal positional encoding -> fp32 + bf16 copies
// ---------------------------------------------------------------------------
__global__ __launch_bounds__(256)
void posenc_kernel(const float* __restrict__ x, float* __restrict__ outF,
                   unsigned short* __restrict__ outB, int S, int D)
{
  const int rowIdx = blockIdx.x;       // b*S + s
  const int s = rowIdx % S;
  const float* xr = x + (size_t)rowIdx * D;
  float* of = outF + (size_t)rowIdx * D;
  unsigned short* ob = outB + (size_t)rowIdx * D;
  for (int d = threadIdx.x; d < D; d += 256) {
    float e     = (2.0f * (float)(d >> 1)) / (float)D;
    float denom = __powf(1000.0f, e);
    float arg   = (float)s / denom;
    float pe    = (d & 1) ? __cosf(arg) : __sinf(arg);
    float v = xr[d] + pe;
    of[d] = v;
    ob[d] = f2bf(v);
  }
}

// ---------------------------------------------------------------------------
// Weight convert + transpose: W f32 [K,N] -> Wt bf16 [N,K]
// ---------------------------------------------------------------------------
__global__ __launch_bounds__(256)
void wconv_kernel(const float* __restrict__ W, unsigned short* __restrict__ Wt,
                  int K, int N)
{
  int idx = blockIdx.x * 256 + threadIdx.x;
  if (idx >= K * N) return;
  int k = idx / N, n = idx - k * N;
  Wt[(size_t)n * K + k] = f2bf(W[idx]);
}

// ---------------------------------------------------------------------------
// Per-token channel attention: for each (b,s,h), aff[q,k]=q[q]*k[k], causal
// mask over the 64-channel axis, softmax, av[q]=sum_k w[q,k] v[k].
// 4 head-tokens per 256-thread block, one thread per q row.
// ---------------------------------------------------------------------------
#define CC 64
__global__ __launch_bounds__(256)
void attn_kernel(const float* __restrict__ Q, const float* __restrict__ Kv,
                 const float* __restrict__ V, unsigned short* __restrict__ avB,
                 int NH, int HID)
{
  __shared__ float kk[4][CC];
  __shared__ float vv[4][CC];
  const int hl  = threadIdx.x >> 6;
  const int q   = threadIdx.x & 63;
  const int ht  = blockIdx.x * 4 + hl;
  const int tok = ht / NH;
  const int h   = ht - tok * NH;
  const size_t base = (size_t)tok * HID + (size_t)h * CC;
  kk[hl][q] = Kv[base + q];
  vv[hl][q] = V[base + q];
  __syncthreads();
  const float qv = Q[base + q];
  float mx = -3.0e38f;
  for (int j = 0; j <= q; ++j) mx = fmaxf(mx, qv * kk[hl][j]);
  float s = 0.0f, a = 0.0f;
  for (int j = 0; j <= q; ++j) {
    float e = __expf(qv * kk[hl][j] - mx);
    s += e;
    a += e * vv[hl][j];
  }
  avB[base + q] = f2bf(a / s);
}

// ---------------------------------------------------------------------------
// out = LayerNorm(y + res) * gamma + beta   (biased variance, eps 1e-5)
// one 256-thread block per row of D=1024; dual fp32 + bf16 output
// ---------------------------------------------------------------------------
__global__ __launch_bounds__(256)
void residual_ln_kernel(const float* __restrict__ y, const float* __restrict__ res,
                        const float* __restrict__ gamma, const float* __restrict__ beta,
                        float* __restrict__ outF, unsigned short* __restrict__ outB, int D)
{
  __shared__ float s1[256], s2[256];
  const size_t ro = (size_t)blockIdx.x * D;
  float vals[4];
  float sum = 0.0f, sq = 0.0f;
  #pragma unroll
  for (int i = 0; i < 4; ++i) {
    int d = threadIdx.x + i * 256;
    float v = y[ro + d] + res[ro + d];
    vals[i] = v; sum += v; sq += v * v;
  }
  s1[threadIdx.x] = sum; s2[threadIdx.x] = sq;
  __syncthreads();
  for (int off = 128; off > 0; off >>= 1) {
    if (threadIdx.x < off) {
      s1[threadIdx.x] += s1[threadIdx.x + off];
      s2[threadIdx.x] += s2[threadIdx.x + off];
    }
    __syncthreads();
  }
  const float mu   = s1[0] / (float)D;
  const float var  = s2[0] / (float)D - mu * mu;
  const float rstd = rsqrtf(var + 1e-5f);
  #pragma unroll
  for (int i = 0; i < 4; ++i) {
    int d = threadIdx.x + i * 256;
    float v = (vals[i] - mu) * rstd * gamma[d] + beta[d];
    outF[ro + d] = v;
    outB[ro + d] = f2bf(v);
  }
}

// ---------------------------------------------------------------------------
extern "C" void kernel_launch(void* const* d_in, const int* in_sizes, int n_in,
                              void* d_out, int out_size, void* d_ws, size_t ws_size,
                              hipStream_t stream)
{
  (void)in_sizes; (void)n_in; (void)out_size; (void)ws_size;

  const int Bb = 2, Ss = 2048, Dd = 1024, HIDl = 1024, NHl = 16, Ll = 4;
  const int Mrows = Bb * Ss;                 // 4096
  const size_t MD = (size_t)Mrows * Dd;

  const float* x     = (const float*)d_in[0];
  const float* Wq    = (const float*)d_in[1];
  const float* bq    = (const float*)d_in[2];
  const float* Wk    = (const float*)d_in[3];
  const float* bk    = (const float*)d_in[4];
  const float* Wv    = (const float*)d_in[5];
  const float* bv    = (const float*)d_in[6];
  const float* Wo    = (const float*)d_in[7];
  const float* bo    = (const float*)d_in[8];
  const float* W1    = (const float*)d_in[9];
  const float* b1    = (const float*)d_in[10];
  const float* W2    = (const float*)d_in[11];
  const float* b2    = (const float*)d_in[12];
  const float* gamma = (const float*)d_in[13];
  const float* beta  = (const float*)d_in[14];

  char* ws = (char*)d_ws;
  size_t off = 0;
  auto alloc = [&](size_t bytes) -> char* {
    char* p = ws + off;
    off += (bytes + 255) & ~(size_t)255;
    return p;
  };
  float*          xcur = (float*)alloc(MD * 4);
  float*          x2   = (float*)alloc(MD * 4);
  float*          ybuf = (float*)alloc(MD * 4);
  float*          qb   = (float*)alloc(MD * 4);
  float*          kb   = (float*)alloc(MD * 4);
  float*          vb   = (float*)alloc(MD * 4);
  unsigned short* xbf  = (unsigned short*)alloc(MD * 2);
  unsigned short* x2bf = (unsigned short*)alloc(MD * 2);
  unsigned short* hbf  = (unsigned short*)alloc(MD * 2);
  unsigned short* avbf = (unsigned short*)alloc(MD * 2);
  unsigned short* wt   = (unsigned short*)alloc((size_t)Dd * HIDl * 2);

  dim3 gGemm(HIDl / BN, Mrows / BM);         // (8, 16)
  const int wcBlocks = (Dd * HIDl) / 256;

  posenc_kernel<<<Mrows, 256, 0, stream>>>(x, xcur, xbf, Ss, Dd);

  for (int l = 0; l < Ll; ++l) {
    const size_t wo = (size_t)l * Dd * HIDl;

    // Q = (x @ Wq + bq) / sqrt(C)
    wconv_kernel<<<wcBlocks, 256, 0, stream>>>(Wq + wo, wt, Dd, HIDl);
    gemm_bf16_wmma<false, true, false><<<gGemm, TB, 0, stream>>>(
        xbf, wt, bq + l * HIDl, 0.125f, qb, nullptr, Mrows, HIDl, Dd);
    // K
    wconv_kernel<<<wcBlocks, 256, 0, stream>>>(Wk + wo, wt, Dd, HIDl);
    gemm_bf16_wmma<false, true, false><<<gGemm, TB, 0, stream>>>(
        xbf, wt, bk + l * HIDl, 1.0f, kb, nullptr, Mrows, HIDl, Dd);
    // V
    wconv_kernel<<<wcBlocks, 256, 0, stream>>>(Wv + wo, wt, Dd, HIDl);
    gemm_bf16_wmma<false, true, false><<<gGemm, TB, 0, stream>>>(
        xbf, wt, bv + l * HIDl, 1.0f, vb, nullptr, Mrows, HIDl, Dd);

    // channel-softmax attention -> av (bf16)
    attn_kernel<<<(Mrows * NHl) / 4, 256, 0, stream>>>(qb, kb, vb, avbf, NHl, HIDl);

    // attn_out = av @ Wo + bo
    wconv_kernel<<<wcBlocks, 256, 0, stream>>>(Wo + wo, wt, HIDl, Dd);
    gemm_bf16_wmma<false, true, false><<<gGemm, TB, 0, stream>>>(
        avbf, wt, bo + l * Dd, 1.0f, ybuf, nullptr, Mrows, Dd, HIDl);

    // x2 = LN(attn_out + x)
    residual_ln_kernel<<<Mrows, 256, 0, stream>>>(
        ybuf, xcur, gamma + l * Dd, beta + l * Dd, x2, x2bf, Dd);

    // h = relu(x2 @ W1 + b1)   (bf16 only)
    wconv_kernel<<<wcBlocks, 256, 0, stream>>>(W1 + wo, wt, Dd, HIDl);
    gemm_bf16_wmma<true, false, true><<<gGemm, TB, 0, stream>>>(
        x2bf, wt, b1 + l * HIDl, 1.0f, nullptr, hbf, Mrows, HIDl, Dd);

    // y = h @ W2 + b2
    wconv_kernel<<<wcBlocks, 256, 0, stream>>>(W2 + wo, wt, HIDl, Dd);
    gemm_bf16_wmma<false, true, false><<<gGemm, TB, 0, stream>>>(
        hbf, wt, b2 + l * Dd, 1.0f, ybuf, nullptr, Mrows, Dd, HIDl);

    // x = LN(y + x2)  -> feeds next layer (fp32 + bf16)
    residual_ln_kernel<<<Mrows, 256, 0, stream>>>(
        ybuf, x2, gamma + l * Dd, beta + l * Dd, xcur, xbf, Dd);
  }

  hipMemcpyAsync(d_out, xcur, MD * sizeof(float), hipMemcpyDeviceToDevice, stream);
}
